// TorchFovea_17154099380619
// MI455X (gfx1250) — matrix operand order repaired and checked by reading, not compiled, not run on hardware
//
#include <hip/hip_runtime.h>

// ---------------------------------------------------------------------------
// TorchFovea on MI455X (gfx1250): bandwidth-bound Laplacian-pyramid foveation.
// 5x5 binomial blurs run through V_WMMA_F32_16X16X32_F16 (separable blur as
// two banded 16x16x32 matmuls per tile). Interior tiles are fetched with the
// Tensor Data Mover (tensor_load_to_lds + s_wait_tensorcnt); border tiles use
// a manual reflect-padded loader. Everything else is fused VALU work.
// ---------------------------------------------------------------------------

typedef __attribute__((ext_vector_type(16))) _Float16 v16h;
typedef __attribute__((ext_vector_type(8)))  _Float16 v8h;
typedef __attribute__((ext_vector_type(4)))  _Float16 v4h;
typedef __attribute__((ext_vector_type(8)))  float    v8f;
typedef __attribute__((ext_vector_type(4)))  unsigned int u32x4;
typedef __attribute__((ext_vector_type(4)))  int      i32x4;
typedef __attribute__((ext_vector_type(8)))  int      i32x8;

#if defined(__has_builtin)
#if __has_builtin(__builtin_amdgcn_tensor_load_to_lds) && \
    __has_builtin(__builtin_amdgcn_s_wait_tensorcnt)
#define HAVE_TDM 1
#endif
#endif
#ifndef HAVE_TDM
#define HAVE_TDM 0
#endif

__device__ __forceinline__ int refl(int p, int n) {
  if (p < 0) p = -p;
  if (p >= n) p = 2 * n - 2 - p;
  return p;
}

// binomial kernel taps {1,4,6,4,1}/16, branchless
__device__ __forceinline__ float k1v(int i) {
  return (i == 2) ? 0.375f : ((i == 1 || i == 3) ? 0.25f : 0.0625f);
}
__device__ __forceinline__ float bandw(int d) {   // w[d] for d in [0,5) else 0
  return (d < 0 || d > 4) ? 0.0f : k1v(d);
}

// ---------------------------------------------------------------------------
// Base foveation filter: f[y,x] = exp(-((x-xc)^2 + (y-yc)^2) / (2*sigma^2)).
// (g/g.max() is identity: the grid contains d==0.)
// ---------------------------------------------------------------------------
__global__ void gen_filter_kernel(float* __restrict__ F, int fh, int fw,
                                  int yc, int xc, float inv_s) {
  int idx = blockIdx.x * blockDim.x + threadIdx.x;
  if (idx >= fh * fw) return;
  int x = idx % fw, y = idx / fw;
  float ex = (float)(x - xc), ey = (float)(y - yc);
  F[idx] = __expf(-(ex * ex + ey * ey) * inv_s);
}

// ---------------------------------------------------------------------------
// pyrDown: 5x5 binomial blur (reflect pad) + drop odd rows/cols.
// One wave32 per 16x16 blurred tile -> 8x8 decimated outputs.
//   H pass:  T(20x16) = In(20x32, cols 20..31 zero) x Bh(32x16 banded)
//   V pass:  O(16x16) = Wv(16x32 banded) x T(32x16, rows 20..31 zero)
// Input tile staged as packed f32 in LDS (TDM layout); T staged transposed
// (s_tT[col][row], 80B col stride) so D-writes/B-reads are b128 DS ops.
// ---------------------------------------------------------------------------
__global__ void pyrdown_wmma_kernel(const float* __restrict__ src,
                                    float* __restrict__ dst,
                                    int Hs, int Ws, int Hd, int Wd, int tilesX) {
  __shared__ float    s_flat[416];     // packed 20x20 tile (+pad, TDM layout)
  __shared__ _Float16 s_tT[16][40];    // transposed H-blur temp: [col][row]

  const int lane = threadIdx.x;                 // wave32
  const int tile = blockIdx.x;
  const int ch   = blockIdx.y;
  const int ty = tile / tilesX, tx = tile - ty * tilesX;
  const int r0 = ty * 16, c0 = tx * 16;
  const float* s = src + (size_t)ch * Hs * Ws;
  float*       d = dst + (size_t)ch * Hd * Wd;

  // zero the K-pad rows (20..31) of the vertical-pass B operand once
  for (int i = lane; i < 16 * 12; i += 32)
    s_tT[i / 12][20 + (i % 12)] = (_Float16)0.0f;

  const bool interior = (r0 >= 2) && (c0 >= 2) &&
                        (r0 + 17 < Hs) && (c0 + 17 < Ws);
#if HAVE_TDM
  if (interior) {
    // Tensor Data Mover: 2D tile 20x20 f32, row stride Ws, packed into LDS.
    const float* gbase = s + (size_t)(r0 - 2) * Ws + (c0 - 2);
    unsigned long long ga = (unsigned long long)(uintptr_t)gbase;
    unsigned ldsa = (unsigned)(uintptr_t)&s_flat[0];
    u32x4 g0;
    g0[0] = 1u;                                         // count=1 (valid D#)
    g0[1] = ldsa;                                       // lds_addr
    g0[2] = (unsigned)(ga & 0xFFFFFFFFu);               // global_addr[31:0]
    g0[3] = (unsigned)((ga >> 32) & 0x01FFFFFFu) | (2u << 30); // [56:32]|type=2
    i32x8 g1;
    g1[0] = (int)(2u << 16);                            // data_size=4B, mask=0
    g1[1] = (int)(20u << 16);                           // tensor_dim0 = 20
    g1[2] = (int)(20u << 16);                           // tensor_dim1 = 20
    g1[3] = (int)(20u << 16);                           // tile_dim0 = 20
    g1[4] = (int)(20u);                                 // tile_dim1=20, dim2=0
    g1[5] = (int)(unsigned)Ws;                          // dim0_stride[31:0]
    g1[6] = 0;                                          // stride hi bits
    g1[7] = 0;
    i32x4 z4 = {0, 0, 0, 0};
#if __clang_major__ >= 23
    i32x8 z8 = {0, 0, 0, 0, 0, 0, 0, 0};
    __builtin_amdgcn_tensor_load_to_lds(g0, g1, z4, z4, z8, 0);
#else
    __builtin_amdgcn_tensor_load_to_lds(g0, g1, z4, z4, 0);
#endif
    __builtin_amdgcn_s_wait_tensorcnt(0);
  } else
#endif
  {
    for (int i = lane; i < 400; i += 32) {
      int rr = i / 20, cc = i - rr * 20;
      int gy = refl(r0 - 2 + rr, Hs);
      int gx = refl(c0 - 2 + cc, Ws);
      s_flat[i] = s[(size_t)gy * Ws + gx];
    }
  }
  __syncthreads();

  const int nm = lane & 15;            // M for A-matrix rows / N for B,C,D cols
  const int hi = (lane >> 4) & 1;      // lane half-group

  // banded horizontal B: Bh[k][n] = w[k-n]   (B layout: K = h + (h<8?0:8) + 8*hi)
  v16h bh;
#pragma unroll
  for (int h = 0; h < 16; ++h)
    bh[h] = (_Float16)bandw((h + ((h < 8) ? 0 : 8) + hi * 8) - nm);

  // horizontal pass over row blocks [0..15] and [4..19]
#pragma unroll
  for (int blk = 0; blk < 2; ++blk) {
    const int off = blk * 4;
    // A layout: K = h + (h<8?0:8) + 8*hi ; rows = off + nm
    const float* rowp = &s_flat[(off + nm) * 20];
    const float4 vA = *(const float4*)(rowp + hi * 8);        // k: hi*8+0..3
    const float4 vB = *(const float4*)(rowp + hi * 8 + 4);    // k: hi*8+4..7
    const float4 vC = *(const float4*)(rowp + 16 + hi * 8);   // k: 16+hi*8..+3
    v16h a;
    a[0] = (_Float16)vA.x;  a[1] = (_Float16)vA.y;
    a[2] = (_Float16)vA.z;  a[3] = (_Float16)vA.w;
    a[4] = (_Float16)vB.x;  a[5] = (_Float16)vB.y;
    a[6] = (_Float16)vB.z;  a[7] = (_Float16)vB.w;
    // k = 16+hi*8+j : valid (<20) only for hi==0
    a[8]  = hi ? (_Float16)0.0f : (_Float16)vC.x;
    a[9]  = hi ? (_Float16)0.0f : (_Float16)vC.y;
    a[10] = hi ? (_Float16)0.0f : (_Float16)vC.z;
    a[11] = hi ? (_Float16)0.0f : (_Float16)vC.w;
    // k = 20+hi*8+j : always >= 20 -> zero
    a[12] = a[13] = a[14] = a[15] = (_Float16)0.0f;

    v8f acc = {};
    acc = __builtin_amdgcn_wmma_f32_16x16x32_f16(
        /*neg_a=*/false, a, /*neg_b=*/false, bh,
        /*c_mod=*/(short)0, acc, /*reuse_a=*/false, /*reuse_b=*/false);

    if (blk == 0) {
      // D rows hi*8 .. hi*8+7 -> contiguous halves in transposed temp: b128
      v8h t;
#pragma unroll
      for (int r = 0; r < 8; ++r) t[r] = (_Float16)acc[r];
      *(v8h*)&s_tT[nm][hi * 8] = t;
    } else if (hi) {
      // only rows 16..19 are new (acc[4..7] of the hi half-group)
      v4h t;
#pragma unroll
      for (int r = 0; r < 4; ++r) t[r] = (_Float16)acc[4 + r];
      *(v4h*)&s_tT[nm][16] = t;
    }
  }
  __syncthreads();

  // vertical pass: banded A, T as B (unconditional b128 reads, pad pre-zeroed)
  v16h av, bt;
  const v8h b0 = *(const v8h*)&s_tT[nm][hi * 16];
  const v8h b1 = *(const v8h*)&s_tT[nm][hi * 16 + 8];
#pragma unroll
  for (int h = 0; h < 16; ++h) {
    av[h] = (_Float16)bandw((h + hi * 16) - nm);
    bt[h] = (h < 8) ? b0[h] : b1[h - 8];
  }
  v8f o = {};
  o = __builtin_amdgcn_wmma_f32_16x16x32_f16(false, av, false, bt,
                                             (short)0, o, false, false);

  // decimate: keep even (gy, gx) of the blurred tile
#pragma unroll
  for (int r = 0; r < 8; ++r) {
    int gy = r0 + r + hi * 8;
    int gx = c0 + nm;
    if (((gy | gx) & 1) == 0) {
      int dy = gy >> 1, dx = gx >> 1;
      if (dy < Hd && dx < Wd) d[(size_t)dy * Wd + dx] = o[r];
    }
  }
}

// ---------------------------------------------------------------------------
// pyrUp (zero-insert + 4*K blur, reflect pad) fused with resize_nearest,
// sampled at one output pixel. Branchless parity masking.
// ---------------------------------------------------------------------------
__device__ __forceinline__ float pyrup_at(const float* __restrict__ src,
                                          int Hn, int Wn, int y, int x,
                                          int H, int W) {
  int ry = (int)(((long long)y * (2 * Hn)) / H);   // resize_nearest: floor
  int rx = (int)(((long long)x * (2 * Wn)) / W);
  float acc = 0.0f;
#pragma unroll
  for (int dy = -2; dy <= 2; ++dy) {
    int p = refl(ry + dy, 2 * Hn);
    float my = (p & 1) ? 0.0f : (4.0f * k1v(dy + 2));
#pragma unroll
    for (int dx = -2; dx <= 2; ++dx) {
      int q = refl(rx + dx, 2 * Wn);
      float m = (q & 1) ? 0.0f : my * k1v(dx + 2);
      acc += m * src[(size_t)(p >> 1) * Wn + (q >> 1)];
    }
  }
  return acc;
}

// ---------------------------------------------------------------------------
// Level combine: fov_i = pyrup(fov_{i+1}) + (cur_i - pyrup(dn_{i+1})) * crop_i
// (for i == LEVEL-1, fov_{i+1} == dn_{i+1} and the two pyrups coincide)
// ---------------------------------------------------------------------------
__global__ void combine_kernel(const float* __restrict__ cur,
                               const float* __restrict__ dnext,
                               const float* __restrict__ fovnext,
                               const float* __restrict__ filt,
                               const float* __restrict__ fix,
                               float* __restrict__ out,
                               int BC, int C, int H, int W, int Hn, int Wn,
                               int fh, int fw, float inv2i, int sameNext) {
  int idx = blockIdx.x * blockDim.x + threadIdx.x;
  if (idx >= BC * H * W) return;
  int x  = idx % W;
  int y  = (idx / W) % H;
  int bc = idx / (W * H);
  int b  = bc / C;

  // per-sample crop origin: x1 = int(fw/2 - fix_x / 2^i)  (positive -> trunc)
  float fxv = fix[2 * b + 0] * inv2i;
  float fyv = fix[2 * b + 1] * inv2i;
  int x1 = (int)(0.5f * (float)fw - fxv);
  int y1 = (int)(0.5f * (float)fh - fyv);
  const float* gp = &filt[(size_t)(y1 + y) * fw + (x1 + x)];
  __builtin_prefetch(gp, 0, 0);                 // global_prefetch_b8

  const float* fn = fovnext + (size_t)bc * Hn * Wn;
  float pf = pyrup_at(fn, Hn, Wn, y, x, H, W);
  float pu = pf;
  if (!sameNext) {
    const float* dn = dnext + (size_t)bc * Hn * Wn;
    pu = pyrup_at(dn, Hn, Wn, y, x, H, W);
  }
  out[idx] = pf + (cur[idx] - pu) * (*gp);
}

// ---------------------------------------------------------------------------
// Host-side launcher
// ---------------------------------------------------------------------------
static void launch_pyrdown(const float* src, float* dst, int C,
                           int Hs, int Ws, int Hd, int Wd, hipStream_t st) {
  int tilesX = (Wd + 7) / 8;
  int tilesY = (Hd + 7) / 8;
  dim3 grid((unsigned)(tilesX * tilesY), (unsigned)C);
  pyrdown_wmma_kernel<<<grid, dim3(32), 0, st>>>(src, dst, Hs, Ws, Hd, Wd, tilesX);
}

static void launch_combine(const float* cur, const float* dn, const float* fov,
                           const float* filt, const float* fix, float* out,
                           int BC, int C, int H, int W, int Hn, int Wn,
                           int fh, int fw, float inv2i, int same, hipStream_t st) {
  int total = BC * H * W;
  combine_kernel<<<(total + 255) / 256, 256, 0, st>>>(
      cur, dn, fov, filt, fix, out, BC, C, H, W, Hn, Wn, fh, fw, inv2i, same);
}

extern "C" void kernel_launch(void* const* d_in, const int* in_sizes, int n_in,
                              void* d_out, int out_size, void* d_ws, size_t ws_size,
                              hipStream_t stream) {
  (void)in_sizes; (void)n_in; (void)out_size; (void)ws_size;
  const float* images = (const float*)d_in[0];   // (32,3,360,640)
  const float* fix    = (const float*)d_in[1];   // (32,2) -> (x,y)
  float* out = (float*)d_out;
  float* ws  = (float*)d_ws;

  const int B = 32, C = 3, BC = B * C;
  // level dims
  const int H0 = 360, W0 = 640, H1 = 180, W1 = 320, H2 = 90, W2 = 160;
  const int H3 = 45,  W3 = 80,  H4 = 23,  W4 = 40,  H5 = 12, W5 = 20;
  // filter dims (level 0..4)
  const int f0h = 719, f0w = 1279, f1h = 360, f1w = 640, f2h = 180, f2w = 320;
  const int f3h = 90,  f3w = 160,  f4h = 45,  f4w = 80;

  size_t off = 0;
  float* F0  = ws + off; off += (size_t)f0h * f0w;
  float* F1  = ws + off; off += (size_t)f1h * f1w;
  float* F2  = ws + off; off += (size_t)f2h * f2w;
  float* F3  = ws + off; off += (size_t)f3h * f3w;
  float* F4  = ws + off; off += (size_t)f4h * f4w;
  float* dn1 = ws + off; off += (size_t)BC * H1 * W1;
  float* dn2 = ws + off; off += (size_t)BC * H2 * W2;
  float* dn3 = ws + off; off += (size_t)BC * H3 * W3;
  float* dn4 = ws + off; off += (size_t)BC * H4 * W4;
  float* dn5 = ws + off; off += (size_t)BC * H5 * W5;
  float* fv4 = ws + off; off += (size_t)BC * H4 * W4;
  float* fv3 = ws + off; off += (size_t)BC * H3 * W3;
  float* fv2 = ws + off; off += (size_t)BC * H2 * W2;
  float* fv1 = ws + off; off += (size_t)BC * H1 * W1;

  // 1) base Gaussian filter (normalization is identity: grid contains d==0)
  {
    int n = f0h * f0w;
    gen_filter_kernel<<<(n + 255) / 256, 256, 0, stream>>>(
        F0, f0h, f0w, /*yc=*/360, /*xc=*/640, /*inv_s=*/1.0f / (2.0f * 71.0f * 71.0f));
  }
  // 2) filter pyramid (WMMA pyrDown, C=1)
  launch_pyrdown(F0, F1, 1, f0h, f0w, f1h, f1w, stream);
  launch_pyrdown(F1, F2, 1, f1h, f1w, f2h, f2w, stream);
  launch_pyrdown(F2, F3, 1, f2h, f2w, f3h, f3w, stream);
  launch_pyrdown(F3, F4, 1, f3h, f3w, f4h, f4w, stream);

  // 3) image pyramid (WMMA pyrDown, C=96)
  launch_pyrdown(images, dn1, BC, H0, W0, H1, W1, stream);
  launch_pyrdown(dn1,    dn2, BC, H1, W1, H2, W2, stream);
  launch_pyrdown(dn2,    dn3, BC, H2, W2, H3, W3, stream);
  launch_pyrdown(dn3,    dn4, BC, H3, W3, H4, W4, stream);
  launch_pyrdown(dn4,    dn5, BC, H4, W4, H5, W5, stream);

  // 4) foveated reconstruction, level 4 -> 0
  launch_combine(dn4,    dn5, dn5, F4, fix, fv4, BC, C, H4, W4, H5, W5, f4h, f4w, 0.0625f, 1, stream);
  launch_combine(dn3,    dn4, fv4, F3, fix, fv3, BC, C, H3, W3, H4, W4, f3h, f3w, 0.125f,  0, stream);
  launch_combine(dn2,    dn3, fv3, F2, fix, fv2, BC, C, H2, W2, H3, W3, f2h, f2w, 0.25f,   0, stream);
  launch_combine(dn1,    dn2, fv2, F1, fix, fv1, BC, C, H1, W1, H2, W2, f1h, f1w, 0.5f,    0, stream);
  launch_combine(images, dn1, fv1, F0, fix, out, BC, C, H0, W0, H1, W1, f0h, f0w, 1.0f,    0, stream);
}
